// EdgeMLPDecoder_62191126446521
// MI455X (gfx1250) — matrix operand order repaired and checked by reading, not compiled, not run on hardware
//
#include <hip/hip_runtime.h>
#include <hip/hip_bf16.h>

typedef __bf16 bf16_t;
typedef __attribute__((ext_vector_type(16))) __bf16 v16bf;
typedef __attribute__((ext_vector_type(8)))  float  v8f;

#define DCH 128          // out_channels (z feature dim)
#define K2  256          // 2*DCH, GEMM K
#define HID 128          // hidden dim, GEMM N
#define LDS_PITCH 272    // bf16 elems per W1T row: 256 + 16 pad (row start 544B -> 32B aligned)
#define TILES_PER_BLOCK 8

__global__ __launch_bounds__(256) void edge_mlp_decoder_kernel(
    const float* __restrict__ z,          // [N_NODES, 128]
    const long long* __restrict__ eli,    // [2, E] int64
    const float* __restrict__ W1,         // [256, 128] row-major
    const float* __restrict__ b1,         // [128]
    const float* __restrict__ W2,         // [128]
    const float* __restrict__ b2,         // [1]
    float* __restrict__ out,              // [E]
    int E)
{
    __shared__ bf16_t sW1T[HID * LDS_PITCH];  // transposed: sW1T[n][k] = bf16(W1[k][n])
    __shared__ float  sB1[HID];
    __shared__ float  sW2[HID];

    const int tid = threadIdx.x;

    // Stage W1 (transposed, bf16) + b1 + W2 into LDS. Global reads are coalesced
    // (i is k-major over W1); LDS write scatter happens once per block.
    for (int i = tid; i < K2 * HID; i += 256) {
        const int k = i >> 7;      // 0..255
        const int n = i & 127;     // 0..127
        sW1T[n * LDS_PITCH + k] = (bf16_t)W1[i];
    }
    for (int i = tid; i < HID; i += 256) {
        sB1[i] = b1[i];
        sW2[i] = W2[i];
    }
    __syncthreads();

    const int wave = tid >> 5;
    const int lane = tid & 31;
    const int m    = lane & 15;    // row (edge) within 16-edge tile for A; N column for B/C
    const int hi   = lane >> 4;    // lane half selector
    const float b2v = b2[0];

    const long long* __restrict__ srcIdx = eli;
    const long long* __restrict__ dstIdx = eli + E;

    for (int t = 0; t < TILES_PER_BLOCK; ++t) {
        const int edgeBase = (blockIdx.x * TILES_PER_BLOCK + t) * 128 + wave * 16;
        if (edgeBase >= E) break;          // wave-uniform: EXEC stays full

        int e = edgeBase + m;
        if (e >= E) e = E - 1;             // clamp (padding lanes compute garbage, never stored)
        const int srcRow = (int)srcIdx[e];
        const int dstRow = (int)dstIdx[e];

        // ---- Load A fragments (16x32 bf16 per K-step, 8 steps = K=256) ----
        // Layout: lanes 0-15 (M=lane): elems 0-15 -> K=kb+0..15
        //         lanes 16-31        : elems 0-15 -> K=kb+16..31
        v16bf a[8];
        const int offA = hi ? 16 : 0;
        #pragma unroll
        for (int k = 0; k < 8; ++k) {
            const float* row = z + (size_t)((k < 4) ? srcRow : dstRow) * DCH
                                 + ((k & 3) * 32 + offA);
            const float4 c0 = *(const float4*)(row +  0);
            const float4 c1 = *(const float4*)(row +  4);
            const float4 c2 = *(const float4*)(row +  8);
            const float4 c3 = *(const float4*)(row + 12);
            v16bf av;
            av[0]=(bf16_t)c0.x;  av[1]=(bf16_t)c0.y;  av[2]=(bf16_t)c0.z;  av[3]=(bf16_t)c0.w;
            av[4]=(bf16_t)c1.x;  av[5]=(bf16_t)c1.y;  av[6]=(bf16_t)c1.z;  av[7]=(bf16_t)c1.w;
            av[8]=(bf16_t)c2.x;  av[9]=(bf16_t)c2.y;  av[10]=(bf16_t)c2.z; av[11]=(bf16_t)c2.w;
            av[12]=(bf16_t)c3.x; av[13]=(bf16_t)c3.y; av[14]=(bf16_t)c3.z; av[15]=(bf16_t)c3.w;
            a[k] = av;
        }

        float p[8];
        #pragma unroll
        for (int r = 0; r < 8; ++r) p[r] = 0.0f;

        // ---- N-tile loop: GEMM (K=256) + fused bias/ReLU/W2-dot per tile ----
        // B layout (32x16 bf16): lanes 0-15: N=lane,   elems 0-15 -> K=kb+0..15
        //                        lanes16-31: N=lane-16, elems 0-15 -> K=kb+16..31
        const int kOffB = hi ? 16 : 0;
        #pragma unroll
        for (int n = 0; n < 8; ++n) {
            const int col = n * 16 + m;
            const bf16_t* wrow = &sW1T[col * LDS_PITCH + kOffB];
            v8f c = {};
            #pragma unroll
            for (int k = 0; k < 8; ++k) {
                const v16bf bfrag = *(const v16bf*)(wrow + k * 32);  // 32B aligned
                c = __builtin_amdgcn_wmma_f32_16x16x32_bf16(
                        /*neg_a=*/false, a[k], /*neg_b=*/false, bfrag,
                        /*c_mod=*/(short)0, c, /*reuse_a=*/false, /*reuse_b=*/false);
            }
            // C layout: elem r -> M = r + 8*hi, N = m. Fuse bias + ReLU + W2 dot.
            const float b1v = sB1[col];
            const float w2v = sW2[col];
            #pragma unroll
            for (int r = 0; r < 8; ++r) {
                float hval = c[r] + b1v;
                hval = hval > 0.0f ? hval : 0.0f;
                p[r] += hval * w2v;
            }
        }

        // ---- Reduce across the 16 N-lanes of each half, then store ----
        #pragma unroll
        for (int r = 0; r < 8; ++r) {
            float v = p[r];
            v += __shfl_xor(v, 1);
            v += __shfl_xor(v, 2);
            v += __shfl_xor(v, 4);
            v += __shfl_xor(v, 8);
            p[r] = v;
        }
        if (m == 0) {  // lanes 0 (edges M=0..7) and 16 (edges M=8..15)
            const int base = edgeBase + hi * 8;
            #pragma unroll
            for (int r = 0; r < 8; ++r) {
                const int eo = base + r;
                if (eo < E) out[eo] = p[r] + b2v;
            }
        }
    }
}

extern "C" void kernel_launch(void* const* d_in, const int* in_sizes, int n_in,
                              void* d_out, int out_size, void* d_ws, size_t ws_size,
                              hipStream_t stream) {
    const float*     z   = (const float*)d_in[0];
    const long long* eli = (const long long*)d_in[1];   // int64 [2, E]
    const float*     W1  = (const float*)d_in[2];
    const float*     b1  = (const float*)d_in[3];
    const float*     W2  = (const float*)d_in[4];
    const float*     b2  = (const float*)d_in[5];
    float* out = (float*)d_out;

    const int E = in_sizes[1] / 2;
    const int edgesPerBlock = TILES_PER_BLOCK * 128;
    const int grid = (E + edgesPerBlock - 1) / edgesPerBlock;

    edge_mlp_decoder_kernel<<<grid, 256, 0, stream>>>(z, eli, W1, b1, W2, b2, out, E);
}
